// SwinTransformerBlock_16904991277396
// MI455X (gfx1250) — matrix operand order repaired and checked by reading, not compiled, not run on hardware
//
#include <hip/hip_runtime.h>

typedef __attribute__((ext_vector_type(16))) _Float16 v16h;
typedef __attribute__((ext_vector_type(8)))  _Float16 v8h;
typedef __attribute__((ext_vector_type(8)))  float    v8f;

#define Bn    32
#define Hh    56
#define Wwid  56
#define Cc    192
#define NHn   6
#define WSn   7
#define SSn   3
#define Nn    49
#define HDn   32
#define HIDn  768
#define NWIN  64
#define TOK   100352          // Bn*NWIN*Nn
#define QSCALE 0.17677669529663687f

// ---------------------------------------------------------------------------
// WMMA helpers: v_wmma_f32_16x16x32_f16, D = A(16x32) * B(32x16) + C
// ---------------------------------------------------------------------------
__device__ inline v8f wmma32(v16h a, v16h b, v8f c) {
  return __builtin_amdgcn_wmma_f32_16x16x32_f16(false, a, false, b, (short)0, c,
                                                false, false);
}

// A fragment (16x32 f16, row-major source, ld in elements).
// lane<16: row=lane, k = {0..7, 16..23}; lane>=16: row=lane-16, k={8..15,24..31}
__device__ inline v16h ld_a(const _Float16* p0, int ld, int lane) {
  int row = lane & 15;
  int kb  = (lane & 16) ? 8 : 0;
  const _Float16* p = p0 + (size_t)row * ld + kb;
  v8h lo = *(const v8h*)(p);
  v8h hi = *(const v8h*)(p + 16);
  v16h r;
#pragma unroll
  for (int i = 0; i < 8; ++i) { r[i] = lo[i]; r[i + 8] = hi[i]; }
  return r;
}

__device__ inline v16h ld_a_clamp(const _Float16* base, int ld, int lane,
                                  int row0, int rmax) {
  int row = row0 + (lane & 15); if (row > rmax) row = rmax;
  int kb  = (lane & 16) ? 8 : 0;
  const _Float16* p = base + (size_t)row * ld + kb;
  v8h lo = *(const v8h*)(p);
  v8h hi = *(const v8h*)(p + 16);
  v16h r;
#pragma unroll
  for (int i = 0; i < 8; ++i) { r[i] = lo[i]; r[i + 8] = hi[i]; }
  return r;
}

// B fragment (32x16 KxN) built from row-major W[n][k]: lane holds column n,
// k = {0..7, 16..23} for lane<16, k = {8..15, 24..31} for lane>=16.
__device__ inline v16h ld_b(const _Float16* p0, int ld, int lane) {
  int col = lane & 15;
  int kb  = (lane & 16) ? 8 : 0;
  const _Float16* p = p0 + (size_t)col * ld + kb;
  v8h lo = *(const v8h*)(p);
  v8h hi = *(const v8h*)(p + 16);
  v16h r;
#pragma unroll
  for (int i = 0; i < 8; ++i) { r[i] = lo[i]; r[i + 8] = hi[i]; }
  return r;
}

__device__ inline v16h ld_b_clamp(const _Float16* base, int ld, int lane,
                                  int col0, int cmax) {
  int col = col0 + (lane & 15); if (col > cmax) col = cmax;
  int kb  = (lane & 16) ? 8 : 0;
  const _Float16* p = base + (size_t)col * ld + kb;
  v8h lo = *(const v8h*)(p);
  v8h hi = *(const v8h*)(p + 16);
  v16h r;
#pragma unroll
  for (int i = 0; i < 8; ++i) { r[i] = lo[i]; r[i + 8] = hi[i]; }
  return r;
}

// ---------------------------------------------------------------------------
// Weight f32 -> f16 conversion (qkv_w | proj_w | fc1_w | fc2_w concatenated)
// ---------------------------------------------------------------------------
__global__ void cvt_weights(const float* __restrict__ a, int na,
                            const float* __restrict__ b, int nb,
                            const float* __restrict__ c, int nc,
                            const float* __restrict__ d, int nd,
                            _Float16* __restrict__ out) {
  int i = blockIdx.x * blockDim.x + threadIdx.x;
  int tot = na + nb + nc + nd;
  if (i >= tot) return;
  float v;
  if (i < na)                v = a[i];
  else if (i < na + nb)      v = b[i - na];
  else if (i < na + nb + nc) v = c[i - na - nb];
  else                       v = d[i - na - nb - nc];
  out[i] = (_Float16)v;
}

// ---------------------------------------------------------------------------
// Precompute combined rel-pos bias + shift mask:
//   BM[head][wi][m][n] = rel_bias_table[relidx(m,n)][head] + mask(wi,m,n)
// 6*64*49*49 floats = 3.7 MB -> lives in L2 for the whole attention pass.
// ---------------------------------------------------------------------------
__global__ void build_bm(const float* __restrict__ rbt, float* __restrict__ BM) {
  int slice = blockIdx.x;                 // head*NWIN + wi
  int head  = slice / NWIN, wi = slice - head * NWIN;
  int wh = wi >> 3, ww = wi & 7;
  for (int e = threadIdx.x; e < Nn * Nn; e += blockDim.x) {
    int m = e / Nn, n = e - m * Nn;
    int i1 = m / 7, j1 = m - 7 * i1, i2 = n / 7, j2 = n - 7 * i2;
    int idx = (i1 - i2 + 6) * 13 + (j1 - j2 + 6);
    float bb = rbt[idx * NHn + head];
    int g1h = wh * 7 + i1, g1w = ww * 7 + j1;
    int g2h = wh * 7 + i2, g2w = ww * 7 + j2;
    int l1 = ((g1h < 49) ? 0 : ((g1h < 53) ? 1 : 2)) * 3 +
             ((g1w < 49) ? 0 : ((g1w < 53) ? 1 : 2));
    int l2 = ((g2h < 49) ? 0 : ((g2h < 53) ? 1 : 2)) * 3 +
             ((g2w < 49) ? 0 : ((g2w < 53) ? 1 : 2));
    BM[(size_t)slice * (Nn * Nn) + e] = bb + ((l1 != l2) ? -100.f : 0.f);
  }
}

// ---------------------------------------------------------------------------
// LayerNorm. gather=1: fused roll(-3,-3) + window partition (dst in window
// order, src gathered from original image).  One wave per token (wave32).
// ---------------------------------------------------------------------------
__global__ void ln_kernel(const float* __restrict__ src, _Float16* __restrict__ dst,
                          const float* __restrict__ gw, const float* __restrict__ gb,
                          int gather) {
  int lane = threadIdx.x & 31;
  int wid  = threadIdx.x >> 5;
  int t    = blockIdx.x * 8 + wid;
  int srow = t;
  if (gather) {
    int w  = t / Nn, n = t - w * Nn;
    int b  = w >> 6, wi = w & 63;
    int wh = wi >> 3, ww = wi & 7;
    int i  = n / WSn, j = n - i * WSn;
    int gh = wh * WSn + i + SSn; if (gh >= Hh)   gh -= Hh;
    int gv = ww * WSn + j + SSn; if (gv >= Wwid) gv -= Wwid;
    srow = b * (Hh * Wwid) + gh * Wwid + gv;
  }
  const float* p = src + (size_t)srow * Cc;
  float v[6], s = 0.f, q = 0.f;
#pragma unroll
  for (int k = 0; k < 6; ++k) { v[k] = p[lane + 32 * k]; s += v[k]; q += v[k] * v[k]; }
#pragma unroll
  for (int m = 16; m; m >>= 1) { s += __shfl_xor(s, m, 32); q += __shfl_xor(q, m, 32); }
  float mean = s * (1.f / Cc);
  float var  = q * (1.f / Cc) - mean * mean;
  float rs   = rsqrtf(var + 1e-5f);
  _Float16* o = dst + (size_t)t * Cc;
#pragma unroll
  for (int k = 0; k < 6; ++k) {
    int c = lane + 32 * k;
    o[c] = (_Float16)((v[k] - mean) * rs * gw[c] + gb[c]);
  }
}

// ---------------------------------------------------------------------------
// Generic WMMA GEMM:  out[M x N] = A[M x K] * W[N x K]^T + bias, + epilogue.
// Block 256 thr = 8 waves; each wave 16x64 tile; grid (M/128, N/64).
// Software-pipelined K loop: next k-step's fragments are loaded before the
// current step's WMMAs so VMEM latency overlaps the matrix pipe.
// ---------------------------------------------------------------------------
enum { M_QKV = 0, M_PROJ = 1, M_FC1 = 2, M_FC2 = 3 };

template <int MODE>
__global__ void gemm_wmma(const _Float16* __restrict__ A, const _Float16* __restrict__ W,
                          const float* __restrict__ bias, int K, int ldo,
                          _Float16* __restrict__ outH, float* __restrict__ outF,
                          const float* __restrict__ resid) {
  int lane = threadIdx.x & 31;
  int wid  = threadIdx.x >> 5;
  int m0   = blockIdx.x * 128 + wid * 16;
  int n0   = blockIdx.y * 64;
  v8f zero = {};
  v8f acc[4];
#pragma unroll
  for (int t = 0; t < 4; ++t) acc[t] = zero;

  const _Float16* Ab = A + (size_t)m0 * K;
  const int nk = K >> 5;

  // prologue: fragments for k-step 0
  v16h a_cur = ld_a(Ab, K, lane);
  v16h b_cur[4];
#pragma unroll
  for (int t = 0; t < 4; ++t)
    b_cur[t] = ld_b(W + (size_t)(n0 + t * 16) * K, K, lane);

  for (int it = 0; it < nk; ++it) {
    v16h a_nxt, b_nxt[4];
    const bool more = (it + 1) < nk;
    if (more) {
      int kn = (it + 1) << 5;
      a_nxt = ld_a(Ab + kn, K, lane);
#pragma unroll
      for (int t = 0; t < 4; ++t)
        b_nxt[t] = ld_b(W + (size_t)(n0 + t * 16) * K + kn, K, lane);
    }
#pragma unroll
    for (int t = 0; t < 4; ++t) acc[t] = wmma32(a_cur, b_cur[t], acc[t]);
    if (more) {
      a_cur = a_nxt;
#pragma unroll
      for (int t = 0; t < 4; ++t) b_cur[t] = b_nxt[t];
    }
  }

  int col0 = lane & 15, grp = lane >> 4;
#pragma unroll
  for (int r = 0; r < 8; ++r) {
    int row = m0 + grp * 8 + r;
    size_t drow = (size_t)row;
    if (MODE == M_PROJ) {         // window-reverse + roll(+3,+3) scatter
      int w  = row / Nn, n = row - w * Nn;
      int b  = w >> 6, wi = w & 63, wh = wi >> 3, ww = wi & 7;
      int i  = n / WSn, j = n - i * WSn;
      int gh = wh * WSn + i + SSn; if (gh >= Hh)   gh -= Hh;
      int gv = ww * WSn + j + SSn; if (gv >= Wwid) gv -= Wwid;
      drow = (size_t)b * (Hh * Wwid) + gh * Wwid + gv;
    }
#pragma unroll
    for (int t = 0; t < 4; ++t) {
      int col = n0 + t * 16 + col0;
      float v = acc[t][r] + bias[col];
      if (MODE == M_QKV) {
        if (col < Cc) v *= QSCALE;                     // fold softmax scale into q
        outH[(size_t)row * ldo + col] = (_Float16)v;
      } else if (MODE == M_FC1) {
        v = 0.5f * v * (1.f + erff(v * 0.70710678118654752f));   // exact GELU
        outH[(size_t)row * ldo + col] = (_Float16)v;
      } else if (MODE == M_PROJ) {
        outF[drow * Cc + col] = resid[drow * Cc + col] + v;      // x1 = x + h
      } else {                                                   // M_FC2
        outF[(size_t)row * Cc + col] = resid[(size_t)row * Cc + col] + v;
      }
    }
  }
}

// ---------------------------------------------------------------------------
// Fused window attention: one wave per (window, head).
// Strip-mined over 16-row strips (tm): S = q k^T (4 wmma) -> + BM[head][wi]
// (precomputed bias+mask; clamped unconditional loads, L2-resident) ->
// row softmax via shuffles -> P strip to LDS.  Then O = P V (16 wmma).
// ---------------------------------------------------------------------------
__global__ void attn_kernel(const _Float16* __restrict__ QKV,
                            const float* __restrict__ BM,
                            _Float16* __restrict__ AO) {
  __shared__ _Float16 Pl[64 * 72];
  __shared__ _Float16 Vt[32 * 72];
  int lane = threadIdx.x;
  int w = blockIdx.x, head = blockIdx.y;
  int wi = w & 63;
  const _Float16* base = QKV + (size_t)(w * Nn) * 576 + head * HDn;
  const float* bm = BM + (size_t)(head * NWIN + wi) * (Nn * Nn);

  // stage V^T (Vt[d][k]) -- zero-padded k rows so padded K contributes 0
  {
    int d = lane;
    for (int kt = 0; kt < 64; ++kt) {
      float vv = (kt < Nn) ? (float)base[(size_t)kt * 576 + 2 * Cc + d] : 0.f;
      Vt[d * 72 + kt] = (_Float16)vv;
    }
  }

  int col0 = lane & 15, grp = lane >> 4;
  v8f zero = {};

  // column (n) clamped indices -- loads are always in-bounds & unconditional
  int nn_c[4];
  bool nn_ok[4];
#pragma unroll
  for (int tn = 0; tn < 4; ++tn) {
    int nn = tn * 16 + col0;
    nn_ok[tn] = nn < Nn;
    nn_c[tn]  = nn_ok[tn] ? nn : (Nn - 1);
  }

  // k fragments, kept live across all strips (32 VGPRs)
  v16h bk[4];
#pragma unroll
  for (int tn = 0; tn < 4; ++tn)
    bk[tn] = ld_b_clamp(base + Cc, 576, lane, tn * 16, Nn - 1);

#pragma unroll
  for (int tm = 0; tm < 4; ++tm) {          // 16-row strip at a time
    v16h aq = ld_a_clamp(base, 576, lane, tm * 16, Nn - 1);   // q (pre-scaled)
    v8f S[4];
#pragma unroll
    for (int tn = 0; tn < 4; ++tn) S[tn] = wmma32(aq, bk[tn], zero);

#pragma unroll
    for (int r = 0; r < 8; ++r) {
      int mm = tm * 16 + grp * 8 + r;
      bool mok = mm < Nn;
      const float* bmrow = bm + (mok ? mm : (Nn - 1)) * Nn;   // clamped, valid
#pragma unroll
      for (int tn = 0; tn < 4; ++tn) {
        float bv = bmrow[nn_c[tn]];                           // unconditional
        S[tn][r] = (mok && nn_ok[tn]) ? (S[tn][r] + bv) : -3.0e30f;
      }
      float mx = -3.0e30f;
#pragma unroll
      for (int tn = 0; tn < 4; ++tn) mx = fmaxf(mx, S[tn][r]);
#pragma unroll
      for (int msk = 1; msk < 16; msk <<= 1) mx = fmaxf(mx, __shfl_xor(mx, msk, 32));
      float sm = 0.f;
#pragma unroll
      for (int tn = 0; tn < 4; ++tn) {
        float e = __expf(S[tn][r] - mx);
        S[tn][r] = e; sm += e;
      }
#pragma unroll
      for (int msk = 1; msk < 16; msk <<= 1) sm += __shfl_xor(sm, msk, 32);
      float inv = 1.f / sm;
#pragma unroll
      for (int tn = 0; tn < 4; ++tn)
        Pl[(tm * 16 + grp * 8 + r) * 72 + tn * 16 + col0] =
            (_Float16)(S[tn][r] * inv);
    }
  }
  __syncthreads();

  // O = P V   (K = 64 in two 32-steps, N = 32 in two 16-tiles)
  v8f O[4][2];
#pragma unroll
  for (int tm = 0; tm < 4; ++tm) { O[tm][0] = zero; O[tm][1] = zero; }
#pragma unroll
  for (int ks = 0; ks < 2; ++ks) {
    v16h bv[2];
#pragma unroll
    for (int tn = 0; tn < 2; ++tn)
      bv[tn] = ld_b(Vt + (size_t)(tn * 16) * 72 + ks * 32, 72, lane);
#pragma unroll
    for (int tm = 0; tm < 4; ++tm) {
      v16h ap = ld_a(Pl + (size_t)(tm * 16) * 72 + ks * 32, 72, lane);
#pragma unroll
      for (int tn = 0; tn < 2; ++tn) O[tm][tn] = wmma32(ap, bv[tn], O[tm][tn]);
    }
  }

#pragma unroll
  for (int tm = 0; tm < 4; ++tm)
#pragma unroll
    for (int r = 0; r < 8; ++r) {
      int mm = tm * 16 + grp * 8 + r;
      if (mm < Nn) {
#pragma unroll
        for (int tn = 0; tn < 2; ++tn) {
          int d = tn * 16 + col0;
          AO[(size_t)(w * Nn + mm) * Cc + head * HDn + d] = (_Float16)O[tm][tn][r];
        }
      }
    }
}

// ---------------------------------------------------------------------------
extern "C" void kernel_launch(void* const* d_in, const int* in_sizes, int n_in,
                              void* d_out, int out_size, void* d_ws, size_t ws_size,
                              hipStream_t stream) {
  const float* x      = (const float*)d_in[0];
  const float* norm_w = (const float*)d_in[1];
  const float* norm_b = (const float*)d_in[2];
  const float* qkv_w  = (const float*)d_in[3];
  const float* qkv_b  = (const float*)d_in[4];
  const float* rbt    = (const float*)d_in[5];
  const float* proj_w = (const float*)d_in[6];
  const float* proj_b = (const float*)d_in[7];
  const float* fc1_w  = (const float*)d_in[8];
  const float* fc1_b  = (const float*)d_in[9];
  const float* fc2_w  = (const float*)d_in[10];
  const float* fc2_b  = (const float*)d_in[11];
  float* out = (float*)d_out;

  const size_t MB = 1ull << 20;
  char* ws = (char*)d_ws;
  _Float16* WT  = (_Float16*)(ws);             //  0.9 MB : f16 weights
  _Float16* QKV = (_Float16*)(ws + 1 * MB);    //  116 MB : qkv f16 (dead after attn)
  _Float16* H1  = QKV;                         //  154 MB : fc1 hidden f16 (reuse)
  _Float16* XW  = (_Float16*)(ws + 160 * MB);  // 38.5 MB : LN1 windows f16
  _Float16* AO  = XW;                          //          attn out f16 (reuse)
  float*    X1  = (float*)(ws + 200 * MB);     //   77 MB : x1 f32
  _Float16* XN2 = (_Float16*)(ws + 278 * MB);  // 38.5 MB : LN2 f16
  float*    BM  = (float*)(ws + 318 * MB);     //  3.7 MB : bias+mask table
  (void)in_sizes; (void)n_in; (void)out_size; (void)ws_size;

  _Float16* Wqkv  = WT;
  _Float16* Wproj = WT + 110592;
  _Float16* Wfc1  = WT + 147456;
  _Float16* Wfc2  = WT + 294912;

  // 1) weights -> f16 ; combined bias+mask table
  cvt_weights<<<1728, 256, 0, stream>>>(qkv_w, 110592, proj_w, 36864,
                                        fc1_w, 147456, fc2_w, 147456, WT);
  build_bm<<<NHn * NWIN, 256, 0, stream>>>(rbt, BM);
  // 2) LN1 + roll(-3,-3) + window partition
  ln_kernel<<<TOK / 8, 256, 0, stream>>>(x, XW, norm_w, norm_b, 1);
  // 3) QKV GEMM (+bias, q pre-scaled)
  gemm_wmma<M_QKV><<<dim3(TOK / 128, 9), 256, 0, stream>>>(
      XW, Wqkv, qkv_b, Cc, 576, QKV, nullptr, nullptr);
  // 4) fused windowed attention
  attn_kernel<<<dim3(2048, NHn), 32, 0, stream>>>(QKV, BM, AO);
  // 5) proj GEMM + window reverse + roll(+3,+3) + residual  -> x1 (f32)
  gemm_wmma<M_PROJ><<<dim3(TOK / 128, 3), 256, 0, stream>>>(
      AO, Wproj, proj_b, Cc, 0, nullptr, X1, x);
  // 6) LN2
  ln_kernel<<<TOK / 8, 256, 0, stream>>>(X1, XN2, norm_w, norm_b, 0);
  // 7) FC1 + GELU
  gemm_wmma<M_FC1><<<dim3(TOK / 128, 12), 256, 0, stream>>>(
      XN2, Wfc1, fc1_b, Cc, HIDn, H1, nullptr, nullptr);
  // 8) FC2 + residual -> out
  gemm_wmma<M_FC2><<<dim3(TOK / 128, 3), 256, 0, stream>>>(
      H1, Wfc2, fc2_b, HIDn, 0, nullptr, out, X1);
}
